// DIEN_49331994362301
// MI455X (gfx1250) — matrix operand order repaired and checked by reading, not compiled, not run on hardware
//
#include <hip/hip_runtime.h>

typedef _Float16 h16;
typedef __attribute__((ext_vector_type(16))) _Float16 v16h;
typedef __attribute__((ext_vector_type(8)))  _Float16 v8h;
typedef __attribute__((ext_vector_type(4)))  _Float16 v4h;
typedef __attribute__((ext_vector_type(8)))  float    v8f;
typedef __attribute__((ext_vector_type(4)))  float    v4f;

// ---------------------------------------------------------------------------
// WMMA fragment helpers (wave32, V_WMMA_F32_16X16X32_F16)
//
// A (16x32, MxK), 16-bit: lane%16 = M, hid = lane/16.
//   halfs 0..7  -> K = kbase + 8*hid + {0..7}
//   halfs 8..15 -> K = kbase + 8*hid + {16..23}
// B (32x16, KxN), 16-bit: lane%16 = N, hid = lane/16 (same K order as A):
//   halfs 0..7  -> K = kbase + 8*hid + {0..7}
//   halfs 8..15 -> K = kbase + 8*hid + {16..23}
// C/D (16x16 f32): lane%16 = N, VGPR r -> M = r + 8*hid.
// ---------------------------------------------------------------------------
__device__ __forceinline__ v16h fragA_ld(const h16* rowptr, int kbase, int hid) {
  const h16* p = rowptr + kbase + hid * 8;
  v8h lo = *(const v8h*)(p);
  v8h hi = *(const v8h*)(p + 16);
  v16h r;
#pragma unroll
  for (int i = 0; i < 8; ++i) { r[i] = lo[i]; r[i + 8] = hi[i]; }
  return r;
}

__device__ __forceinline__ v16h fragB_ld(const h16* rowptr, int kbase, int hid) {
  const h16* p = rowptr + kbase + hid * 8;
  v8h lo = *(const v8h*)(p);
  v8h hi = *(const v8h*)(p + 16);
  v16h r;
#pragma unroll
  for (int i = 0; i < 8; ++i) { r[i] = lo[i]; r[i + 8] = hi[i]; }
  return r;
}

__device__ __forceinline__ v8f wmma16(v16h a, v16h b, v8f c) {
  return __builtin_amdgcn_wmma_f32_16x16x32_f16(false, a, false, b, (short)0, c,
                                                false, false);
}

__device__ __forceinline__ float sigmoidf_(float x) {
  return 1.0f / (1.0f + __expf(-x));
}

// ---------------------------------------------------------------------------
// Problem constants
// ---------------------------------------------------------------------------
static constexpr int Bb = 1024, Ss = 200;
static constexpr int MROWS = Bb * Ss;             // 204800
static constexpr int FCIN = 608;                  // 586 padded to 19*32

// ---------------------------------------------------------------------------
// f32 -> f16 weight convert with zero padding
// ---------------------------------------------------------------------------
__global__ void k_cvt_pad(const float* __restrict__ src, h16* __restrict__ dst,
                          int srows, int scols, int drows, int dcols) {
  int i = blockIdx.x * blockDim.x + threadIdx.x;
  int n = drows * dcols;
  if (i >= n) return;
  int r = i / dcols, c = i % dcols;
  float v = (r < srows && c < scols) ? src[r * scols + c] : 0.0f;
  dst[i] = (h16)v;
}

// ---------------------------------------------------------------------------
// hist = [item_table[hist_items] | cat_table[hist_cats]]  -> f16 (M,192)
// ---------------------------------------------------------------------------
__global__ void k_gather_hist(const int* __restrict__ hist_items,
                              const int* __restrict__ hist_cats,
                              const float* __restrict__ item_table,
                              const float* __restrict__ cat_table,
                              h16* __restrict__ hist_h) {
  int i = blockIdx.x * blockDim.x + threadIdx.x;   // over MROWS*48
  if (i >= MROWS * 48) return;
  int row = i / 48, g = i % 48;
  v4f v;
  if (g < 32) v = *(const v4f*)(item_table + (size_t)hist_items[row] * 128 + g * 4);
  else        v = *(const v4f*)(cat_table + (size_t)hist_cats[row] * 64 + (g - 32) * 4);
  v4h o;
#pragma unroll
  for (int j = 0; j < 4; ++j) o[j] = (h16)v[j];
  *(v4h*)(hist_h + (size_t)row * 192 + g * 4) = o;
}

__global__ void k_gather_tgt(const int* __restrict__ article_ids,
                             const int* __restrict__ category_ids,
                             const float* __restrict__ item_table,
                             const float* __restrict__ cat_table,
                             h16* __restrict__ tgt_h) {
  int i = blockIdx.x * blockDim.x + threadIdx.x;   // over B*48
  if (i >= Bb * 48) return;
  int b = i / 48, g = i % 48;
  v4f v;
  if (g < 32) v = *(const v4f*)(item_table + (size_t)article_ids[b] * 128 + g * 4);
  else        v = *(const v4f*)(cat_table + (size_t)category_ids[b] * 64 + (g - 32) * 4);
  v4h o;
#pragma unroll
  for (int j = 0; j < 4; ++j) o[j] = (h16)v[j];
  *(v4h*)(tgt_h + (size_t)b * 192 + g * 4) = o;
}

// ---------------------------------------------------------------------------
// Generic GEMM: C(M,N) = A(M,K) @ W(N,K)^T + bias, f16 in / f32 out.
// A fragments hoisted into registers once per M-tile, reused across NTPW
// N-tiles. launch_bounds(128,1) -> full register budget, no spills.
// ---------------------------------------------------------------------------
template <int KTILES, int NTPW>
__global__ __launch_bounds__(128, 1) void k_gemm_t(
    const h16* __restrict__ A, int lda,
    const h16* __restrict__ W, int ldw,
    const float* __restrict__ bias,
    float* __restrict__ C, int ldc, int ngroups) {
  int lane = threadIdx.x & 31, hid = lane >> 4, l16 = lane & 15;
  int wave = (blockIdx.x * blockDim.x + threadIdx.x) >> 5;
  int mtile = wave / ngroups, ngrp = wave % ngroups;
  const h16* arow = A + (size_t)(mtile * 16 + l16) * lda;
  v16h af[KTILES];
#pragma unroll
  for (int kt = 0; kt < KTILES; ++kt) af[kt] = fragA_ld(arow, kt * 32, hid);
  for (int nt = 0; nt < NTPW; ++nt) {
    int n0 = (ngrp * NTPW + nt) * 16;
    const h16* brow = W + (size_t)(n0 + l16) * ldw;
    float bv = bias ? bias[n0 + l16] : 0.0f;
    v8f c = {};
#pragma unroll
    for (int kt = 0; kt < KTILES; ++kt)
      c = wmma16(af[kt], fragB_ld(brow, kt * 32, hid), c);
    float* cp = C + (size_t)(mtile * 16 + hid * 8) * ldc + n0 + l16;
#pragma unroll
    for (int r = 0; r < 8; ++r) cp[(size_t)r * ldc] = c[r] + bv;
  }
}

// ---------------------------------------------------------------------------
// GRU scan: 64 blocks x 16 batch rows, 8 waves. Hidden state lives in LDS;
// recurrent weights (3 N-tiles x 4 K-chunks per wave = 96 VGPRs) are hoisted
// into registers before the 200-step loop. launch_bounds(256,1) gives the
// register budget to keep them resident (round-2 build spilled to scratch).
// Bias is added at the LDS store, keeping accumulator init = 0 (short live
// ranges instead of 24 永-live bias registers).
// ---------------------------------------------------------------------------
__global__ __launch_bounds__(256, 1) void k_gru_scan(
    const h16* __restrict__ Whh_h,   // 384x128 f16
    const float* __restrict__ gx,    // (M,384)
    const float* __restrict__ bhh,
    const int* __restrict__ seq_lens,
    float* __restrict__ interests,
    h16* __restrict__ interests_h) {
  __shared__ h16 h_h[16 * 128];
  __shared__ float h_f[16 * 128];
  __shared__ float gh[16 * 384];
  __shared__ int lens[16];
  int tid = threadIdx.x, lane = tid & 31, hid = lane >> 4, l16 = lane & 15,
      wave = tid >> 5;
  int b0 = blockIdx.x * 16;
  // hoist loop-invariant weight fragments + bias
  v16h wf[3][4];
  float wb[3];
#pragma unroll
  for (int nt = 0; nt < 3; ++nt) {
    int n0 = (wave * 3 + nt) * 16;
    wb[nt] = bhh[n0 + l16];
    const h16* brow = Whh_h + (size_t)(n0 + l16) * 128;
#pragma unroll
    for (int kt = 0; kt < 4; ++kt) wf[nt][kt] = fragB_ld(brow, kt * 32, hid);
  }
  for (int i = tid; i < 16 * 128; i += 256) { h_f[i] = 0.0f; h_h[i] = (h16)0.0f; }
  if (tid < 16) { int L = seq_lens[b0 + tid]; lens[tid] = L < 1 ? 1 : L; }
  __syncthreads();
  for (int t = 0; t < Ss; ++t) {
    const h16* arow = h_h + l16 * 128;
    v16h af[4];
#pragma unroll
    for (int kt = 0; kt < 4; ++kt) af[kt] = fragA_ld(arow, kt * 32, hid);
#pragma unroll
    for (int nt = 0; nt < 3; ++nt) {
      int n0 = (wave * 3 + nt) * 16;
      v8f c = {};
#pragma unroll
      for (int kt = 0; kt < 4; ++kt) c = wmma16(af[kt], wf[nt][kt], c);
#pragma unroll
      for (int r = 0; r < 8; ++r)
        gh[(r + hid * 8) * 384 + n0 + l16] = c[r] + wb[nt];
    }
    __syncthreads();
    for (int i = tid; i < 2048; i += 256) {
      int row = i >> 7, j = i & 127;
      int b = b0 + row;
      size_t gxo = ((size_t)b * Ss + t) * 384;
      float rg = sigmoidf_(gx[gxo + j] + gh[row * 384 + j]);
      float zg = sigmoidf_(gx[gxo + 128 + j] + gh[row * 384 + 128 + j]);
      float ng = tanhf(gx[gxo + 256 + j] + rg * gh[row * 384 + 256 + j]);
      if (t + 1 < Ss) {  // prefetch next step's pre-activations (L2 -> L0)
        __builtin_prefetch(&gx[gxo + 384 + j], 0, 3);
        __builtin_prefetch(&gx[gxo + 384 + 128 + j], 0, 3);
        __builtin_prefetch(&gx[gxo + 384 + 256 + j], 0, 3);
      }
      float hv = h_f[row * 128 + j];
      float hn = (1.0f - zg) * ng + zg * hv;
      bool m = t < lens[row];
      float hkeep = m ? hn : hv;
      float iv = m ? hn : 0.0f;
      h_f[row * 128 + j] = hkeep;
      h_h[row * 128 + j] = (h16)hkeep;
      size_t io = ((size_t)b * Ss + t) * 128 + j;
      interests[io] = iv;
      interests_h[io] = (h16)iv;
    }
    __syncthreads();
  }
}

// ---------------------------------------------------------------------------
// Fused attention MLP: relu(x@W1^T+b1) -> relu(@W2p^T+b2) -> sigmoid(@W3+b3)
// x = [interests(128) | tgt(192)].  One wave per 16-row M-tile.
// ---------------------------------------------------------------------------
__global__ __launch_bounds__(128, 1) void k_attention(
    const h16* __restrict__ interests_h,
    const h16* __restrict__ tgt_h,
    const h16* __restrict__ W1h, const float* __restrict__ b1,
    const h16* __restrict__ W2p, const float* __restrict__ b2,
    const float* __restrict__ W3, const float* __restrict__ b3,
    float* __restrict__ att) {
  __shared__ h16 a1h[4 * 16 * 96];
  __shared__ float a2f[4 * 16 * 48];
  int tid = threadIdx.x, lane = tid & 31, hid = lane >> 4, l16 = lane & 15,
      wave = tid >> 5;
  for (int i = tid; i < 4 * 16 * 96; i += 128) a1h[i] = (h16)0.0f;
  __syncthreads();
  int mtile = blockIdx.x * 4 + wave;
  int grow = mtile * 16 + l16;
  int brow = grow / Ss;
  const h16* irow = interests_h + (size_t)grow * 128;
  const h16* trow = tgt_h + (size_t)brow * 192;
  // stage 1: K=320 (10 chunks), N=80 (5 tiles)
  for (int nt = 0; nt < 5; ++nt) {
    int n0 = nt * 16;
    const h16* brw = W1h + (size_t)(n0 + l16) * 320;
    float bv = b1[n0 + l16];
    v8f c = {};
    for (int kt = 0; kt < 10; ++kt) {
      int kb = kt * 32;
      v16h a = (kb < 128) ? fragA_ld(irow, kb, hid) : fragA_ld(trow, kb - 128, hid);
      v16h b = fragB_ld(brw, kb, hid);
      c = wmma16(a, b, c);
    }
#pragma unroll
    for (int r = 0; r < 8; ++r) {
      float v = c[r] + bv; v = v > 0.0f ? v : 0.0f;
      a1h[(wave * 16 + r + hid * 8) * 96 + n0 + l16] = (h16)v;
    }
  }
  asm volatile("s_wait_dscnt 0" ::: "memory");  // same-wave LDS RAW
  // stage 2: K=96 padded (3 chunks), N=48 padded (3 tiles)
  const h16* a1row = a1h + (size_t)(wave * 16 + l16) * 96;
  for (int nt = 0; nt < 3; ++nt) {
    int ncol = nt * 16 + l16;
    const h16* brw = W2p + (size_t)ncol * 96;
    float bv = (ncol < 40) ? b2[ncol] : 0.0f;
    v8f c = {};
#pragma unroll
    for (int kt = 0; kt < 3; ++kt)
      c = wmma16(fragA_ld(a1row, kt * 32, hid), fragB_ld(brw, kt * 32, hid), c);
#pragma unroll
    for (int r = 0; r < 8; ++r) {
      float v = c[r] + bv; v = v > 0.0f ? v : 0.0f;
      a2f[(wave * 16 + r + hid * 8) * 48 + ncol] = v;
    }
  }
  asm volatile("s_wait_dscnt 0" ::: "memory");
  // stage 3: 40 -> 1, lanes 0..15 each handle one row
  if (hid == 0) {
    float s = b3[0];
    const float* ar = a2f + (wave * 16 + l16) * 48;
    for (int j = 0; j < 40; ++j) s += ar[j] * W3[j];
    att[mtile * 16 + l16] = sigmoidf_(s);
  }
}

// ---------------------------------------------------------------------------
// AUGRU scan: per step  u = att * sigmoid(gr + h@Whr^T + br),
//                       ht = tanh(gn + (h*u)@Whh^T + bh),  h = (1-u)h + u*ht
// Both recurrent weight matrices register-resident across the 200 steps.
// ---------------------------------------------------------------------------
__global__ __launch_bounds__(256, 1) void k_augru(
    const h16* __restrict__ Whr_h, const h16* __restrict__ Whh_h,
    const float* __restrict__ br, const float* __restrict__ bh,
    const float* __restrict__ gr_gn,  // (M,256)
    const float* __restrict__ att,    // (M)
    float* __restrict__ evolved) {    // (B,128)
  __shared__ h16 h_h[16 * 128];
  __shared__ h16 hu_h[16 * 128];
  __shared__ float h_f[16 * 128];
  __shared__ float gbuf[16 * 128];
  __shared__ float ubuf[16 * 128];
  int tid = threadIdx.x, lane = tid & 31, hid = lane >> 4, l16 = lane & 15,
      wave = tid >> 5;
  int b0 = blockIdx.x * 16;
  int n0 = wave * 16;
  v16h wrf[4], whf[4];
  {
    const h16* brw_r = Whr_h + (size_t)(n0 + l16) * 128;
    const h16* brw_h = Whh_h + (size_t)(n0 + l16) * 128;
#pragma unroll
    for (int kt = 0; kt < 4; ++kt) {
      wrf[kt] = fragB_ld(brw_r, kt * 32, hid);
      whf[kt] = fragB_ld(brw_h, kt * 32, hid);
    }
  }
  float bvr = br[n0 + l16], bvh = bh[n0 + l16];
  for (int i = tid; i < 2048; i += 256) { h_f[i] = 0.0f; h_h[i] = (h16)0.0f; }
  __syncthreads();
  for (int t = 0; t < Ss; ++t) {
    {  // phase A: h @ Whr^T + br
      const h16* arow = h_h + l16 * 128;
      v8f c = {};
#pragma unroll
      for (int kt = 0; kt < 4; ++kt)
        c = wmma16(fragA_ld(arow, kt * 32, hid), wrf[kt], c);
#pragma unroll
      for (int r = 0; r < 8; ++r)
        gbuf[(r + hid * 8) * 128 + n0 + l16] = c[r] + bvr;
    }
    __syncthreads();
    for (int i = tid; i < 2048; i += 256) {
      int row = i >> 7, j = i & 127;
      int b = b0 + row;
      size_t go = ((size_t)b * Ss + t) * 256;
      float u = sigmoidf_(gr_gn[go + j] + gbuf[i]);
      u *= att[(size_t)b * Ss + t];
      if (t + 1 < Ss) {  // prefetch next step's gr/gn
        __builtin_prefetch(&gr_gn[go + 256 + j], 0, 3);
        __builtin_prefetch(&gr_gn[go + 256 + 128 + j], 0, 3);
      }
      ubuf[i] = u;
      hu_h[i] = (h16)(h_f[i] * u);
    }
    __syncthreads();
    {  // phase C: (h*u) @ Whh^T + bh
      const h16* arow = hu_h + l16 * 128;
      v8f c = {};
#pragma unroll
      for (int kt = 0; kt < 4; ++kt)
        c = wmma16(fragA_ld(arow, kt * 32, hid), whf[kt], c);
#pragma unroll
      for (int r = 0; r < 8; ++r)
        gbuf[(r + hid * 8) * 128 + n0 + l16] = c[r] + bvh;
    }
    __syncthreads();
    for (int i = tid; i < 2048; i += 256) {
      int row = i >> 7, j = i & 127;
      int b = b0 + row;
      size_t go = ((size_t)b * Ss + t) * 256;
      float ht = tanhf(gr_gn[go + 128 + j] + gbuf[i]);
      float u = ubuf[i];
      float hn = (1.0f - u) * h_f[i] + u * ht;
      h_f[i] = hn;
      h_h[i] = (h16)hn;
    }
    __syncthreads();
  }
  for (int i = tid; i < 2048; i += 256) {
    int row = i >> 7, j = i & 127;
    evolved[(size_t)(b0 + row) * 128 + j] = h_f[i];
  }
}

// ---------------------------------------------------------------------------
// Build fc input: [user(128)|item(128)|cat(64)|evolved(128)|int_last(128)|feat(10)|0-pad]
// ---------------------------------------------------------------------------
__global__ void k_concat(const int* __restrict__ user_ids,
                         const int* __restrict__ article_ids,
                         const int* __restrict__ category_ids,
                         const float* __restrict__ user_table,
                         const float* __restrict__ item_table,
                         const float* __restrict__ cat_table,
                         const float* __restrict__ evolved,
                         const float* __restrict__ interests,
                         const float* __restrict__ features,
                         h16* __restrict__ fc_in) {
  int i = blockIdx.x * blockDim.x + threadIdx.x;
  if (i >= Bb * FCIN) return;
  int b = i / FCIN, j = i % FCIN;
  float v;
  if (j < 128)      v = user_table[(size_t)user_ids[b] * 128 + j];
  else if (j < 256) v = item_table[(size_t)article_ids[b] * 128 + (j - 128)];
  else if (j < 320) v = cat_table[(size_t)category_ids[b] * 64 + (j - 256)];
  else if (j < 448) v = evolved[(size_t)b * 128 + (j - 320)];
  else if (j < 576) v = interests[((size_t)b * Ss + (Ss - 1)) * 128 + (j - 448)];
  else if (j < 586) v = features[b * 10 + (j - 576)];
  else              v = 0.0f;
  fc_in[i] = (h16)v;
}

// ---------------------------------------------------------------------------
// Fused final MLP: 608->256->128->64->1, all WMMA except the last dot.
// One block (8 waves) per 16 batch rows.
// ---------------------------------------------------------------------------
__global__ __launch_bounds__(256, 1) void k_final_mlp(
    const h16* __restrict__ fc_in,
    const h16* __restrict__ W1, const float* __restrict__ b1,
    const h16* __restrict__ W2, const float* __restrict__ b2,
    const h16* __restrict__ W3, const float* __restrict__ b3,
    const float* __restrict__ W4, const float* __restrict__ b4,
    float* __restrict__ out) {
  __shared__ h16 x1h[16 * 256];
  __shared__ h16 x2h[16 * 128];
  __shared__ float x3f[16 * 64];
  int tid = threadIdx.x, lane = tid & 31, hid = lane >> 4, l16 = lane & 15,
      wave = tid >> 5;
  int m0 = blockIdx.x * 16;
  const h16* arow = fc_in + (size_t)(m0 + l16) * FCIN;
  // fc1: K=608 (19 chunks), N=256 (2 tiles per wave)
  for (int nt2 = 0; nt2 < 2; ++nt2) {
    int n0 = (wave * 2 + nt2) * 16;
    const h16* brw = W1 + (size_t)(n0 + l16) * FCIN;
    float bv = b1[n0 + l16];
    v8f c = {};
    for (int kt = 0; kt < 19; ++kt)
      c = wmma16(fragA_ld(arow, kt * 32, hid), fragB_ld(brw, kt * 32, hid), c);
#pragma unroll
    for (int r = 0; r < 8; ++r) {
      float v = c[r] + bv; v = v > 0.0f ? v : 0.0f;
      x1h[(r + hid * 8) * 256 + n0 + l16] = (h16)v;
    }
  }
  __syncthreads();
  // fc2: K=256 (8 chunks), N=128 (1 tile per wave)
  {
    int n0 = wave * 16;
    const h16* a1row = x1h + l16 * 256;
    const h16* brw = W2 + (size_t)(n0 + l16) * 256;
    float bv = b2[n0 + l16];
    v8f c = {};
#pragma unroll
    for (int kt = 0; kt < 8; ++kt)
      c = wmma16(fragA_ld(a1row, kt * 32, hid), fragB_ld(brw, kt * 32, hid), c);
#pragma unroll
    for (int r = 0; r < 8; ++r) {
      float v = c[r] + bv; v = v > 0.0f ? v : 0.0f;
      x2h[(r + hid * 8) * 128 + n0 + l16] = (h16)v;
    }
  }
  __syncthreads();
  // fc3: K=128 (4 chunks), N=64 (waves 0..3)
  if (wave < 4) {
    int n0 = wave * 16;
    const h16* a2row = x2h + l16 * 128;
    const h16* brw = W3 + (size_t)(n0 + l16) * 128;
    float bv = b3[n0 + l16];
    v8f c = {};
#pragma unroll
    for (int kt = 0; kt < 4; ++kt)
      c = wmma16(fragA_ld(a2row, kt * 32, hid), fragB_ld(brw, kt * 32, hid), c);
#pragma unroll
    for (int r = 0; r < 8; ++r) {
      float v = c[r] + bv; v = v > 0.0f ? v : 0.0f;
      x3f[(r + hid * 8) * 64 + n0 + l16] = v;
    }
  }
  __syncthreads();
  // fc4: 64 -> 1
  if (tid < 16) {
    float s = b4[0];
    for (int j = 0; j < 64; ++j) s += x3f[tid * 64 + j] * W4[j];
    out[m0 + tid] = sigmoidf_(s);
  }
}

// ---------------------------------------------------------------------------
// Host launcher
// ---------------------------------------------------------------------------
extern "C" void kernel_launch(void* const* d_in, const int* in_sizes, int n_in,
                              void* d_out, int out_size, void* d_ws, size_t ws_size,
                              hipStream_t stream) {
  (void)in_sizes; (void)n_in; (void)out_size; (void)ws_size;
  const int* user_ids     = (const int*)d_in[0];
  const int* article_ids  = (const int*)d_in[1];
  const int* category_ids = (const int*)d_in[2];
  const int* hist_items   = (const int*)d_in[3];
  const int* hist_cats    = (const int*)d_in[4];
  const int* seq_lens     = (const int*)d_in[5];
  const float* features   = (const float*)d_in[6];
  const float* user_table = (const float*)d_in[7];
  const float* item_table = (const float*)d_in[8];
  const float* cat_table  = (const float*)d_in[9];
  const float* gru_Wih = (const float*)d_in[10];
  const float* gru_Whh = (const float*)d_in[11];
  const float* gru_bih = (const float*)d_in[12];
  const float* gru_bhh = (const float*)d_in[13];
  const float* att_W1  = (const float*)d_in[14];
  const float* att_b1  = (const float*)d_in[15];
  const float* att_W2  = (const float*)d_in[16];
  const float* att_b2  = (const float*)d_in[17];
  const float* att_W3  = (const float*)d_in[18];
  const float* att_b3  = (const float*)d_in[19];
  const float* au_Wir  = (const float*)d_in[20];
  const float* au_Whr  = (const float*)d_in[21];
  const float* au_br   = (const float*)d_in[22];
  const float* au_Wih  = (const float*)d_in[23];
  const float* au_Whh  = (const float*)d_in[24];
  const float* au_bh   = (const float*)d_in[25];
  const float* fc1_W = (const float*)d_in[26];
  const float* fc1_b = (const float*)d_in[27];
  const float* fc2_W = (const float*)d_in[28];
  const float* fc2_b = (const float*)d_in[29];
  const float* fc3_W = (const float*)d_in[30];
  const float* fc3_b = (const float*)d_in[31];
  const float* fc4_W = (const float*)d_in[32];
  const float* fc4_b = (const float*)d_in[33];
  float* out = (float*)d_out;

  char* ws = (char*)d_ws;
  size_t off = 0;
  auto take = [&](size_t bytes) -> char* {
    char* p = ws + off;
    off += (bytes + 255) & ~(size_t)255;
    return p;
  };
  const size_t M = (size_t)MROWS;
  h16*   hist_h      = (h16*)take(M * 192 * 2);
  h16*   tgt_h       = (h16*)take((size_t)Bb * 192 * 2);
  float* gx          = (float*)take(M * 384 * 4);
  float* interests   = (float*)take(M * 128 * 4);
  h16*   interests_h = (h16*)take(M * 128 * 2);
  float* gr_gn       = (float*)take(M * 256 * 4);
  float* att         = (float*)take(M * 4);
  float* evolved     = (float*)take((size_t)Bb * 128 * 4);
  h16*   fc_in       = (h16*)take((size_t)Bb * FCIN * 2);
  h16*   Wih_h   = (h16*)take(384 * 192 * 2);
  h16*   Whh_h   = (h16*)take(384 * 128 * 2);
  h16*   W1h     = (h16*)take(80 * 320 * 2);
  h16*   W2p     = (h16*)take(48 * 96 * 2);
  h16*   auStack = (h16*)take(256 * 128 * 2);   // [Wir;Wih]
  h16*   Whr_h   = (h16*)take(128 * 128 * 2);
  h16*   auWhh_h = (h16*)take(128 * 128 * 2);
  h16*   fc1Wh   = (h16*)take(256 * FCIN * 2);
  h16*   fc2Wh   = (h16*)take(128 * 256 * 2);
  h16*   fc3Wh   = (h16*)take(64 * 128 * 2);

  auto cvt = [&](const float* s, h16* dst, int sr, int sc, int dr, int dc) {
    int n = dr * dc;
    k_cvt_pad<<<(n + 255) / 256, 256, 0, stream>>>(s, dst, sr, sc, dr, dc);
  };
  cvt(gru_Wih, Wih_h, 384, 192, 384, 192);
  cvt(gru_Whh, Whh_h, 384, 128, 384, 128);
  cvt(att_W1, W1h, 80, 320, 80, 320);
  cvt(att_W2, W2p, 40, 80, 48, 96);
  cvt(au_Wir, auStack, 128, 128, 128, 128);
  cvt(au_Wih, auStack + 128 * 128, 128, 128, 128, 128);
  cvt(au_Whr, Whr_h, 128, 128, 128, 128);
  cvt(au_Whh, auWhh_h, 128, 128, 128, 128);
  cvt(fc1_W, fc1Wh, 256, 586, 256, FCIN);
  cvt(fc2_W, fc2Wh, 128, 256, 128, 256);
  cvt(fc3_W, fc3Wh, 64, 128, 64, 128);

  k_gather_hist<<<(int)((M * 48 + 255) / 256), 256, 0, stream>>>(
      hist_items, hist_cats, item_table, cat_table, hist_h);
  k_gather_tgt<<<(Bb * 48 + 255) / 256, 256, 0, stream>>>(
      article_ids, category_ids, item_table, cat_table, tgt_h);

  // gx = hist @ Wih^T + bih : M=204800, K=192 (6 chunks), N=384 (3 groups of 8)
  k_gemm_t<6, 8><<<9600, 128, 0, stream>>>(hist_h, 192, Wih_h, 192, gru_bih,
                                           gx, 384, 3);
  // masked GRU scan -> interests (f32 + f16)
  k_gru_scan<<<64, 256, 0, stream>>>(Whh_h, gx, gru_bhh, seq_lens, interests,
                                     interests_h);
  // gr||gn = interests @ [Wir;Wih]^T : K=128 (4 chunks), N=256 (2 groups of 8)
  k_gemm_t<4, 8><<<6400, 128, 0, stream>>>(interests_h, 128, auStack, 128,
                                           nullptr, gr_gn, 256, 2);
  // attention scores
  k_attention<<<3200, 128, 0, stream>>>(interests_h, tgt_h, W1h, att_b1, W2p,
                                        att_b2, att_W3, att_b3, att);
  // AUGRU scan -> evolved
  k_augru<<<64, 256, 0, stream>>>(Whr_h, auWhh_h, au_br, au_bh, gr_gn, att,
                                  evolved);
  // concat + final MLP
  k_concat<<<(Bb * FCIN + 255) / 256, 256, 0, stream>>>(
      user_ids, article_ids, category_ids, user_table, item_table, cat_table,
      evolved, interests, features, fc_in);
  k_final_mlp<<<64, 256, 0, stream>>>(fc_in, fc1Wh, fc1_b, fc2Wh, fc2_b, fc3Wh,
                                      fc3_b, fc4_W, fc4_b, out);
}
